// InterpretableMultiHeadAttention_75325136437826
// MI455X (gfx1250) — compile-verified, matrix operand order checked by reading, not run on hardware
//
#include <hip/hip_runtime.h>

// ---------------------------------------------------------------------------
// InterpretableMultiHeadAttention on MI455X (gfx1250, wave32, WMMA bf16)
// B=1, S=384, D=512, H=8, DK=64.  Output: [S,S,D] fp32 (302MB) + attn head 7.
// bf16 v_wmma_f32_16x16x32_bf16 GEMMs; the huge final GEMM (147456x512x512)
// fuses A = (1/8)*sum_h attn[h,s,t]*hv[h,t,:] generation into its LDS staging
// so avg[S,S,D] is never materialized (saves ~450MB HBM traffic; kernel then
// sits near the 302MB-output store roofline: ~13us at 23.3 TB/s).
// GEMM config: 128 threads = 4 wave32 waves, block tile 128x128, K-step 64;
// each wave owns a 64x64 block = 4x4 wmma accums -> 16 v_wmma per 32-K step
// fed by 16 ds_load_b128 fragment reads (8 for A, 8 for B).
// ---------------------------------------------------------------------------

typedef __attribute__((ext_vector_type(16))) __bf16 v16bf;
typedef __attribute__((ext_vector_type(8)))  float  v8f;

static __device__ __forceinline__ float bf2f(unsigned short u) {
  union { unsigned int i; float f; } x; x.i = ((unsigned int)u) << 16; return x.f;
}
static __device__ __forceinline__ unsigned short f2bf(float f) {
  union { unsigned int i; float f; } x; x.f = f;
  unsigned int r = x.i + 0x7FFFu + ((x.i >> 16) & 1u);   // round-nearest-even
  return (unsigned short)(r >> 16);
}

union FragBF { uint4 u[2]; v16bf v; };
union Pack8  { unsigned short s[8]; uint4 u; };

// ------------------------- elementwise fp32 -> bf16 ------------------------
__global__ void cvt_f32_bf16(const float* __restrict__ src,
                             unsigned short* __restrict__ dst, int n) {
  int i = blockIdx.x * blockDim.x + threadIdx.x;
  if (i < n) dst[i] = f2bf(src[i]);
}

// ------------------- [K][N] fp32 -> [N][K] bf16 transpose ------------------
// grid: (N/32, K/32, batch), block (32,8). All dims are multiples of 32.
__global__ void transpose_bf16(const float* __restrict__ src,
                               unsigned short* __restrict__ dst, int K, int N) {
  src += (size_t)blockIdx.z * K * N;
  dst += (size_t)blockIdx.z * K * N;
  __shared__ float tile[32][33];
  int n0 = blockIdx.x * 32, k0 = blockIdx.y * 32;
#pragma unroll
  for (int i = 0; i < 4; i++) {
    int k = k0 + threadIdx.y + i * 8;
    tile[threadIdx.y + i * 8][threadIdx.x] = src[(size_t)k * N + n0 + threadIdx.x];
  }
  __syncthreads();
#pragma unroll
  for (int i = 0; i < 4; i++) {
    int n = n0 + threadIdx.y + i * 8;
    dst[(size_t)n * K + k0 + threadIdx.x] = f2bf(tile[threadIdx.x][threadIdx.y + i * 8]);
  }
}

// --------------------------- generic WMMA GEMM -----------------------------
// C[m,n] = alpha * sum_k A[m,k]*Bt[n,k] (+ bias[n]).  A row-major [M,K] bf16,
// Bt row-major [N,K] bf16 (pre-transposed B).  Tile 128x128, K-step 64.
// 128 threads = 4 waves (2x2); each wave: 64x64 block = 4x4 wmma accums.
// A_FUSED: A[r,k] = (1/8)*sum_h attn[h, r/Sdim, r%Sdim] * hv[h, r%Sdim, k].
template <bool A_FUSED, bool OUT_BF16, bool HAS_BIAS>
__global__ __launch_bounds__(128) void wmma_gemm_bf16(
    const unsigned short* __restrict__ A, long long strideA, int lda,
    const unsigned short* __restrict__ Bt, long long strideB, int ldb,
    void* __restrict__ Cv, long long strideC, int ldc,
    const float* __restrict__ bias, long long strideBias,
    int K, float alpha,
    const float* __restrict__ attn, const unsigned short* __restrict__ hvp,
    int Sdim, int ldhv) {
  constexpr int BM = 128, BN = 128, BK = 64, LDS_K = BK + 8;  // 144B row pitch
  __shared__ __align__(16) unsigned short As[BM * LDS_K];
  __shared__ __align__(16) unsigned short Bs[BN * LDS_K];
  __shared__ float wlds[BM][8];   // fused: per-row attn weights * 1/8
  __shared__ int   tlds[BM];      // fused: per-row t index

  const int z = blockIdx.z;
  if constexpr (!A_FUSED) A += (size_t)z * strideA;
  Bt += (size_t)z * strideB;

  const int tid = threadIdx.x, lane = tid & 31, wid = tid >> 5;
  const int wm = wid & 1, wn = wid >> 1;                     // 2 x 2 waves
  const int row0 = blockIdx.y * BM, col0 = blockIdx.x * BN;
  const int mlo = lane & 15, khalf = lane >> 4;
  const int srow = tid >> 3;                // staging row within 16-row group
  const int scol = (tid & 7) * 8;           // staging column (8 bf16 = 16B)

  v8f acc[4][4];
#pragma unroll
  for (int i = 0; i < 4; i++)
#pragma unroll
    for (int j = 0; j < 4; j++)
#pragma unroll
      for (int g = 0; g < 8; g++) acc[i][j][g] = 0.0f;

  // Fused: attn weights depend only on the output row -> build LDS table once.
  if constexpr (A_FUSED) {
    if (tid < BM) {
      int gr = row0 + tid;
      int s_ = gr / Sdim, t_ = gr - s_ * Sdim;
      tlds[tid] = t_;
#pragma unroll
      for (int h = 0; h < 8; h++)
        wlds[tid][h] = attn[((size_t)h * Sdim + s_) * Sdim + t_] * 0.125f;
    }
    __syncthreads();
  }

  for (int k0 = 0; k0 < K; k0 += BK) {
    // ---- stage A tile [128 x 64] into LDS ----
    if constexpr (!A_FUSED) {
#pragma unroll
      for (int p = 0; p < 8; p++) {
        int r = p * 16 + srow;
        uint4 d = *(const uint4*)(A + (size_t)(row0 + r) * lda + (k0 + scol));
        *(uint4*)&As[r * LDS_K + scol] = d;
      }
    } else {
#pragma unroll
      for (int p = 0; p < 8; p++) {
        int r = p * 16 + srow;
        int t_ = tlds[r];
        float f[8];
#pragma unroll
        for (int j = 0; j < 8; j++) f[j] = 0.0f;
#pragma unroll
        for (int h = 0; h < 8; h++) {
          uint4 raw = *(const uint4*)(hvp + ((size_t)h * Sdim + t_) * (size_t)ldhv + (k0 + scol));
          const unsigned short* hb = (const unsigned short*)&raw;
          float w = wlds[r][h];
#pragma unroll
          for (int j = 0; j < 8; j++) f[j] += w * bf2f(hb[j]);
        }
        Pack8 ob;
#pragma unroll
        for (int j = 0; j < 8; j++) ob.s[j] = f2bf(f[j]);
        *(uint4*)&As[r * LDS_K + scol] = ob.u;
      }
    }
    // ---- stage Bt tile [128 x 64] into LDS ----
#pragma unroll
    for (int p = 0; p < 8; p++) {
      int r = p * 16 + srow;
      uint4 d = *(const uint4*)(Bt + (size_t)(col0 + r) * ldb + (k0 + scol));
      *(uint4*)&Bs[r * LDS_K + scol] = d;
    }
    // ---- prefetch next K-chunk lines (gfx1250 global_prefetch_b8) ----
    if (k0 + BK < K && (tid & 7) == 0) {
#pragma unroll
      for (int p = 0; p < 8; p++) {
        int r = p * 16 + srow;
        if constexpr (!A_FUSED)
          __builtin_prefetch(A + (size_t)(row0 + r) * lda + (k0 + BK), 0, 3);
        __builtin_prefetch(Bt + (size_t)(col0 + r) * ldb + (k0 + BK), 0, 3);
      }
    }
    __syncthreads();

    // ---- two 32-deep WMMA steps per stage: 16 wmma / 16 ds_load_b128 ----
#pragma unroll
    for (int kk = 0; kk < BK; kk += 32) {
      v16bf afr[4], bfr[4];
#pragma unroll
      for (int i = 0; i < 4; i++) {
        const unsigned short* rp = &As[(wm * 64 + i * 16 + mlo) * LDS_K];
        FragBF f;
        f.u[0] = *(const uint4*)(rp + kk + khalf * 8);        // K = kh*8 + 0..7
        f.u[1] = *(const uint4*)(rp + kk + 16 + khalf * 8);   // K = 16+kh*8+0..7
        afr[i] = f.v;
      }
#pragma unroll
      for (int j = 0; j < 4; j++) {
        const unsigned short* rp = &Bs[(wn * 64 + j * 16 + mlo) * LDS_K];
        FragBF f;
        f.u[0] = *(const uint4*)(rp + kk + khalf * 8);
        f.u[1] = *(const uint4*)(rp + kk + 16 + khalf * 8);
        bfr[j] = f.v;
      }
#pragma unroll
      for (int i = 0; i < 4; i++)
#pragma unroll
        for (int j = 0; j < 4; j++)
          acc[i][j] = __builtin_amdgcn_wmma_f32_16x16x32_bf16(
              false, afr[i], false, bfr[j], (short)0, acc[i][j], false, false);
    }
    __syncthreads();
  }

  // ---- epilogue: C/D layout VGPR g: M = g + (lane>=16)*8, N = lane&15 ----
#pragma unroll
  for (int i = 0; i < 4; i++) {
#pragma unroll
    for (int j = 0; j < 4; j++) {
      int nn = col0 + wn * 64 + j * 16 + mlo;
      float bb = 0.0f;
      if constexpr (HAS_BIAS) bb = bias[(size_t)z * strideBias + nn];
      int mbase = row0 + wm * 64 + i * 16 + khalf * 8;
#pragma unroll
      for (int g = 0; g < 8; g++) {
        float val = alpha * acc[i][j][g] + bb;
        size_t idx = (size_t)z * strideC + (size_t)(mbase + g) * ldc + nn;
        if constexpr (OUT_BF16) ((unsigned short*)Cv)[idx] = f2bf(val);
        else                    ((float*)Cv)[idx] = val;
      }
    }
  }
}

// ----------------------------- row softmax ---------------------------------
// grid: 8*384 rows of length 384; 128 threads * 3 elems. Head 7 rows are also
// written into the second output region (reference returns attn[-1]).
__global__ __launch_bounds__(128) void softmax_rows(float* __restrict__ sc,
                                                    float* __restrict__ attn_last) {
  int row = blockIdx.x;  // h*384 + s
  float* p = sc + (size_t)row * 384;
  int tid = threadIdx.x;
  float v0 = p[tid], v1 = p[tid + 128], v2 = p[tid + 256];
  __shared__ float red[128];
  red[tid] = fmaxf(v0, fmaxf(v1, v2));
  __syncthreads();
  for (int s = 64; s > 0; s >>= 1) {
    if (tid < s) red[tid] = fmaxf(red[tid], red[tid + s]);
    __syncthreads();
  }
  float m = red[0];
  __syncthreads();
  v0 = __expf(v0 - m); v1 = __expf(v1 - m); v2 = __expf(v2 - m);
  red[tid] = v0 + v1 + v2;
  __syncthreads();
  for (int s = 64; s > 0; s >>= 1) {
    if (tid < s) red[tid] += red[tid + s];
    __syncthreads();
  }
  float inv = 1.0f / red[0];
  v0 *= inv; v1 *= inv; v2 *= inv;
  p[tid] = v0; p[tid + 128] = v1; p[tid + 256] = v2;
  if (row >= 7 * 384) {
    float* q = attn_last + (size_t)(row - 7 * 384) * 384;
    q[tid] = v0; q[tid + 128] = v1; q[tid + 256] = v2;
  }
}

// ------------------------------- launcher ----------------------------------
extern "C" void kernel_launch(void* const* d_in, const int* in_sizes, int n_in,
                              void* d_out, int out_size, void* d_ws, size_t ws_size,
                              hipStream_t stream) {
  (void)in_sizes; (void)n_in; (void)out_size; (void)ws_size;
  constexpr int S = 384, D = 512, H = 8, DK = 64;
  constexpr long long OUT0 = (long long)S * S * D;  // 75,497,472 floats

  const float* x    = (const float*)d_in[0];
  const float* Wq   = (const float*)d_in[1];
  const float* bq   = (const float*)d_in[2];
  const float* Wk   = (const float*)d_in[3];
  const float* bk   = (const float*)d_in[4];
  const float* Wv   = (const float*)d_in[5];
  const float* bv   = (const float*)d_in[6];
  const float* Wq_h = (const float*)d_in[7];
  const float* bq_h = (const float*)d_in[8];
  const float* Wk_h = (const float*)d_in[9];
  const float* bk_h = (const float*)d_in[10];
  const float* Wv_s = (const float*)d_in[11];
  const float* bv_s = (const float*)d_in[12];
  const float* Wo   = (const float*)d_in[13];
  const float* bo   = (const float*)d_in[14];
  float* out = (float*)d_out;

  char* ws = (char*)d_ws;
  size_t off = 0;
  auto alloc = [&](size_t bytes) -> void* {
    void* p = ws + off;
    off += (bytes + 255) & ~(size_t)255;
    return p;
  };
  unsigned short* xb   = (unsigned short*)alloc((size_t)S * D * 2);
  unsigned short* WqT  = (unsigned short*)alloc((size_t)D * D * 2);
  unsigned short* WkT  = (unsigned short*)alloc((size_t)D * D * 2);
  unsigned short* WvT  = (unsigned short*)alloc((size_t)D * D * 2);
  unsigned short* WqhT = (unsigned short*)alloc((size_t)H * D * DK * 2);
  unsigned short* WkhT = (unsigned short*)alloc((size_t)H * D * DK * 2);
  unsigned short* WvsT = (unsigned short*)alloc((size_t)D * DK * 2);
  unsigned short* WoT  = (unsigned short*)alloc((size_t)D * D * 2);
  unsigned short* qb   = (unsigned short*)alloc((size_t)S * D * 2);
  unsigned short* kb   = (unsigned short*)alloc((size_t)S * D * 2);
  unsigned short* vb   = (unsigned short*)alloc((size_t)S * D * 2);
  unsigned short* hq   = (unsigned short*)alloc((size_t)H * S * D * 2);
  unsigned short* hk   = (unsigned short*)alloc((size_t)H * S * D * 2);
  unsigned short* hvb  = (unsigned short*)alloc((size_t)H * S * D * 2);
  float* scores        = (float*)alloc((size_t)H * S * S * 4);

  dim3 gblk(128), tb(32, 8);

  // x -> bf16
  cvt_f32_bf16<<<dim3((S * D + 255) / 256), dim3(256), 0, stream>>>(x, xb, S * D);
  // weight transposes: [K][N] fp32 -> [N][K] bf16
  transpose_bf16<<<dim3(16, 16, 1), tb, 0, stream>>>(Wq, WqT, D, D);
  transpose_bf16<<<dim3(16, 16, 1), tb, 0, stream>>>(Wk, WkT, D, D);
  transpose_bf16<<<dim3(16, 16, 1), tb, 0, stream>>>(Wv, WvT, D, D);
  transpose_bf16<<<dim3(16, 2, 8), tb, 0, stream>>>(Wq_h, WqhT, DK, D);
  transpose_bf16<<<dim3(16, 2, 8), tb, 0, stream>>>(Wk_h, WkhT, DK, D);
  transpose_bf16<<<dim3(16, 2, 1), tb, 0, stream>>>(Wv_s, WvsT, DK, D);
  transpose_bf16<<<dim3(16, 16, 1), tb, 0, stream>>>(Wo, WoT, D, D);

  // q/k/v = x @ W + b   (384x512x512, bf16 out)
  wmma_gemm_bf16<false, true, true><<<dim3(4, 3, 1), gblk, 0, stream>>>(
      xb, 0, D, WqT, 0, D, qb, 0, D, bq, 0, D, 1.0f, nullptr, nullptr, 0, 0);
  wmma_gemm_bf16<false, true, true><<<dim3(4, 3, 1), gblk, 0, stream>>>(
      xb, 0, D, WkT, 0, D, kb, 0, D, bk, 0, D, 1.0f, nullptr, nullptr, 0, 0);
  wmma_gemm_bf16<false, true, true><<<dim3(4, 3, 1), gblk, 0, stream>>>(
      xb, 0, D, WvT, 0, D, vb, 0, D, bv, 0, D, 1.0f, nullptr, nullptr, 0, 0);

  // per-head DK->D projections (batched over heads via blockIdx.z)
  wmma_gemm_bf16<false, true, true><<<dim3(4, 3, 8), gblk, 0, stream>>>(
      qb, DK, D, WqhT, (long long)D * DK, DK, hq, (long long)S * D, D,
      bq_h, D, DK, 1.0f, nullptr, nullptr, 0, 0);
  wmma_gemm_bf16<false, true, true><<<dim3(4, 3, 8), gblk, 0, stream>>>(
      kb, DK, D, WkhT, (long long)D * DK, DK, hk, (long long)S * D, D,
      bk_h, D, DK, 1.0f, nullptr, nullptr, 0, 0);
  wmma_gemm_bf16<false, true, true><<<dim3(4, 3, 8), gblk, 0, stream>>>(
      vb, DK, D, WvsT, 0, DK, hvb, (long long)S * D, D,
      bv_s, 0, DK, 1.0f, nullptr, nullptr, 0, 0);

  // scores[h] = hq[h] @ hk[h]^T / sqrt(64)   (Bt == hk row-major directly)
  wmma_gemm_bf16<false, false, false><<<dim3(3, 3, 8), gblk, 0, stream>>>(
      hq, (long long)S * D, D, hk, (long long)S * D, D,
      scores, (long long)S * S, S, nullptr, 0, D, 0.125f, nullptr, nullptr, 0, 0);

  // softmax rows (in-place); head 7 also -> output tail
  softmax_rows<<<dim3(H * S), dim3(128), 0, stream>>>(scores, out + OUT0);

  // out = [fused (1/8)*sum_h attn*hv] @ Wo + bo   (147456 x 512 x 512)
  wmma_gemm_bf16<true, false, true><<<dim3(4, 1152, 1), gblk, 0, stream>>>(
      nullptr, 0, 0, WoT, 0, D, out, 0, D, bo, 0, D, 1.0f, scores, hvb, S, D);
}
